// WorldModel_31602369364187
// MI455X (gfx1250) — matrix operand order repaired and checked by reading, not compiled, not run on hardware
//
#include <hip/hip_runtime.h>
#include <hip/hip_bf16.h>

typedef __attribute__((ext_vector_type(16))) __bf16          bf16x16;
typedef __attribute__((ext_vector_type(8)))  float           f32x8;
typedef __attribute__((ext_vector_type(8)))  unsigned short  u16x8;
typedef __attribute__((ext_vector_type(16))) unsigned short  u16x16;
typedef __attribute__((ext_vector_type(4)))  unsigned int    u32x4;
typedef __attribute__((ext_vector_type(8)))  int             i32x8;
typedef __attribute__((ext_vector_type(4)))  int             i32x4;

__device__ __forceinline__ unsigned short f2bf(float f) {
    unsigned int u = __builtin_bit_cast(unsigned int, f);
    u += 0x7fffu + ((u >> 16) & 1u);           // round-to-nearest-even
    return (unsigned short)(u >> 16);
}
__device__ __forceinline__ float sigm(float x)   { return 1.0f / (1.0f + __expf(-x)); }
__device__ __forceinline__ float tanh_f(float x) { return 2.0f * sigm(2.0f * x) - 1.0f; }

__device__ __forceinline__ f32x8 zero8() {
    f32x8 v;
#pragma unroll
    for (int j = 0; j < 8; ++j) v[j] = 0.0f;
    return v;
}

// A fragment (16x32 bf16): lane holds row = lane&15, K = (lane>>4)*16 + e
__device__ __forceinline__ bf16x16 frag_a_rowptr(const unsigned short* prow, int lane) {
    const int kb = (lane >> 4) * 16;
    u16x8 lo = *(const u16x8*)(prow + kb);
    u16x8 hi = *(const u16x8*)(prow + kb + 8);
    u16x16 t;
#pragma unroll
    for (int i = 0; i < 8; ++i) { t[i] = lo[i]; t[i + 8] = hi[i]; }
    return __builtin_bit_cast(bf16x16, t);
}
__device__ __forceinline__ bf16x16 frag_a(const unsigned short* base, int ld, int lane) {
    return frag_a_rowptr(base + (size_t)(lane & 15) * ld, lane);
}
// B fragment (32x16 bf16) from row-major W[n,k] (B = W^T): lane col = lane&15,
// K = k + (lane>>4)*16 + e  -> single 32B load per lane
__device__ __forceinline__ bf16x16 frag_b(const unsigned short* w, int ld, int n0, int k, int lane) {
    const int col = lane & 15;
    const int kb  = (lane >> 4) * 16;
    u16x16 v = *(const u16x16*)(w + (size_t)(n0 + col) * ld + k + kb);
    return __builtin_bit_cast(bf16x16, v);
}
__device__ __forceinline__ f32x8 wmma(bf16x16 a, bf16x16 b, f32x8 c) {
    return __builtin_amdgcn_wmma_f32_16x16x32_bf16(false, a, false, b, (short)0, c, false, false);
}

// ---- Tensor Data Mover: 1-D bf16 tile (nelem u16 elements), optional LDS row padding ----
// pad: insert 8 DWORDs (32B) after every 128 DWORDs (512B = one 256-elem bf16 row)
// D# per CDNA5 ISA 8.3/8.4: group0 = {count=1, lds_addr, global_addr, type=2},
// group1 = {data_size=2B, pad cfg, tensor_dim0, tile_dim0}
// This toolchain exposes the 6-arg builtin: (g0, g1, g2, g3, extra, cpol);
// groups 2/3 + extra are unused for a 1-D tile -> zeros.
__device__ __forceinline__ void tdm_load_1d(unsigned lds_off, const void* gptr,
                                            unsigned nelem, int pad) {
    unsigned long long ga = (unsigned long long)(uintptr_t)gptr;
    u32x4 g0;
    g0[0] = 1u;                                       // count=1, user descriptor
    g0[1] = lds_off;                                  // lds_addr (bytes)
    g0[2] = (unsigned)(ga & 0xffffffffu);             // global_addr[31:0]
    g0[3] = (unsigned)((ga >> 32) & 0x01ffffffu) | (2u << 30);  // addr[56:32] | type=2
    i32x8 g1;
    unsigned d0 = (1u << 16);                         // data_size = 1 -> 2 bytes
    if (pad) d0 |= (1u << 20) | (6u << 22) | (7u << 25);  // pad_en, interval=128dw, amount=8dw
    g1[0] = (int)d0;
    g1[1] = (int)((nelem & 0xffffu) << 16);           // tensor_dim0[15:0] in [31:16]
    g1[2] = (int)((nelem >> 16) & 0xffffu);           // tensor_dim0[31:16]; tensor_dim1=0
    g1[3] = (int)((nelem & 0xffffu) << 16);           // tile_dim0 (1-D tile)
    g1[4] = 0; g1[5] = 0; g1[6] = 0; g1[7] = 0;       // tile_dim1/2 = 0, strides = 0
    i32x4 gz4; gz4[0] = gz4[1] = gz4[2] = gz4[3] = 0;
    i32x8 gz8;
#pragma unroll
    for (int i = 0; i < 8; ++i) gz8[i] = 0;
    __builtin_amdgcn_tensor_load_to_lds(g0, g1, gz4, gz4, gz8, 0);
}

// ---------------- weight conversion kernels ----------------
__global__ void k_cvt(const float* __restrict__ s, unsigned short* __restrict__ d, int n) {
    for (int i = blockIdx.x * blockDim.x + threadIdx.x; i < n; i += gridDim.x * blockDim.x)
        d[i] = f2bf(s[i]);
}
__global__ void k_cvt_pad(const float* __restrict__ s, unsigned short* __restrict__ d,
                          int rows, int cols, int ld) {
    int n = rows * ld;
    for (int i = blockIdx.x * blockDim.x + threadIdx.x; i < n; i += gridDim.x * blockDim.x) {
        int r = i / ld, c = i - r * ld;
        d[i] = (c < cols) ? f2bf(s[(size_t)r * cols + c]) : (unsigned short)0;
    }
}

// ---------------- encoder: z = obs @ w_enc^T + b_enc  (15360 x 7056 -> 512), bf16 out ----
__global__ __launch_bounds__(512) void k_encoder(const float* __restrict__ obs,
                                                 const unsigned short* __restrict__ wenc,
                                                 const float* __restrict__ benc,
                                                 unsigned short* __restrict__ zbf) {
    __shared__ __attribute__((aligned(32))) unsigned short so[64 * 80];
    const int tid  = threadIdx.x;
    const int lane = tid & 31;
    const int wid  = tid >> 5;
    const int rt   = wid & 3;
    const int ct0  = (wid >> 2) * 8;
    const int row_base = blockIdx.x * 64;

    f32x8 acc[8];
#pragma unroll
    for (int c = 0; c < 8; ++c) acc[c] = zero8();

    const int lr = tid >> 3;
    const int lj = (tid & 7) * 8;

    for (int kc = 0; kc < 7056; kc += 64) {
        float vals[8];
        if (kc + lj < 7056) {
            const float* src = obs + (size_t)(row_base + lr) * 7056 + kc + lj;
#pragma unroll
            for (int i = 0; i < 8; ++i) vals[i] = src[i];
        } else {
#pragma unroll
            for (int i = 0; i < 8; ++i) vals[i] = 0.0f;
        }
        unsigned short* dst = so + lr * 80 + lj;
#pragma unroll
        for (int i = 0; i < 8; ++i) dst[i] = f2bf(vals[i]);
        if (kc + 64 + lj < 7056)
            __builtin_prefetch(obs + (size_t)(row_base + lr) * 7056 + kc + 64 + lj, 0, 0);
        __syncthreads();

#pragma unroll
        for (int kk = 0; kk < 64; kk += 32) {
            bf16x16 a = frag_a(so + (rt * 16) * 80 + kk, 80, lane);
#pragma unroll
            for (int c = 0; c < 8; ++c)
                acc[c] = wmma(a, frag_b(wenc, 7072, (ct0 + c) * 16, kc + kk, lane), acc[c]);
        }
        __syncthreads();
    }
    const int col   = lane & 15;
    const int rhalf = (lane >> 4) * 8;
#pragma unroll
    for (int c = 0; c < 8; ++c) {
        int n = (ct0 + c) * 16 + col;
        float bias = benc[n];
#pragma unroll
        for (int v = 0; v < 8; ++v) {
            int rg = row_base + rt * 16 + v + rhalf;
            zbf[(size_t)rg * 512 + n] = f2bf(acc[c][v] + bias);
        }
    }
}

// ---------------- sequential GRU scan: 4 blocks (16 batch rows each), 232 steps ----------
__global__ __launch_bounds__(512) void k_gru_scan(const unsigned short* __restrict__ zbf,
                                                  const unsigned short* __restrict__ wih,
                                                  const unsigned short* __restrict__ whh,
                                                  const float* __restrict__ bih,
                                                  const float* __restrict__ bhh,
                                                  float* __restrict__ hof,
                                                  unsigned short* __restrict__ hob) {
    __shared__ __attribute__((aligned(32))) float          hf[16 * 256];
    __shared__ __attribute__((aligned(32))) unsigned short hb[16 * 272];
    const int tid  = threadIdx.x, lane = tid & 31, wid = tid >> 5;
    const int r0   = blockIdx.x * 16;
    const int m0   = wid * 16;
    for (int i = tid; i < 16 * 256; i += 512) hf[i] = 0.0f;
    for (int i = tid; i < 16 * 272; i += 512) hb[i] = 0;
    __syncthreads();

    const int col = lane & 15, rhalf = (lane >> 4) * 8;
    const int m   = m0 + col;
    const float br  = bih[m] + bhh[m];
    const float bz  = bih[256 + m] + bhh[256 + m];
    const float bin = bih[512 + m];
    const float bhn = bhh[512 + m];

    for (int t = 0; t < 232; ++t) {
        f32x8 ar = zero8(), az = zero8(), ain = zero8(), ahn = zero8();
        const unsigned short* zrow = zbf + (size_t)(t * 64 + r0) * 512;
        for (int k = 0; k < 512; k += 32) {
            bf16x16 a = frag_a(zrow + k, 512, lane);
            ar  = wmma(a, frag_b(wih, 512, m0,        k, lane), ar);
            az  = wmma(a, frag_b(wih, 512, 256 + m0,  k, lane), az);
            ain = wmma(a, frag_b(wih, 512, 512 + m0,  k, lane), ain);
        }
#pragma unroll
        for (int k = 0; k < 256; k += 32) {
            bf16x16 a = frag_a(hb + k, 272, lane);
            ar  = wmma(a, frag_b(whh, 256, m0,        k, lane), ar);
            az  = wmma(a, frag_b(whh, 256, 256 + m0,  k, lane), az);
            ahn = wmma(a, frag_b(whh, 256, 512 + m0,  k, lane), ahn);
        }
        float h2v[8];
#pragma unroll
        for (int v = 0; v < 8; ++v) {
            int rl = v + rhalf;
            float r  = sigm(ar[v] + br);
            float zz = sigm(az[v] + bz);
            float nn = tanh_f(ain[v] + bin + r * (ahn[v] + bhn));
            h2v[v] = (1.0f - zz) * nn + zz * hf[rl * 256 + m];
        }
        __syncthreads();
#pragma unroll
        for (int v = 0; v < 8; ++v) {
            int rl = v + rhalf;
            unsigned short hbf = f2bf(h2v[v]);
            hf[rl * 256 + m] = h2v[v];
            hb[rl * 272 + m] = hbf;
            if (t >= 40) {
                size_t orow = (size_t)(t - 40) * 64 + r0 + rl;
                hof[orow * 256 + m] = h2v[v];
                hob[orow * 256 + m] = hbf;
            }
        }
        __syncthreads();
    }
}

// ------- open-loop GRU step: 64-row stripes, B-fragment reuse x4, TDM-staged h ----------
__global__ __launch_bounds__(512) void k_open_gru(int kstep,
                                                  const int* __restrict__ acts,
                                                  const unsigned short* __restrict__ embb,
                                                  const unsigned short* __restrict__ wihc,
                                                  const unsigned short* __restrict__ whhc,
                                                  const float* __restrict__ bihc,
                                                  const float* __restrict__ bhhc,
                                                  const float* __restrict__ hinf,
                                                  const unsigned short* __restrict__ hinb,
                                                  float* __restrict__ houtf,
                                                  unsigned short* __restrict__ houtb) {
    __shared__ __attribute__((aligned(64))) unsigned short hstage[64 * 272];
    const int tid  = threadIdx.x, lane = tid & 31, wid = tid >> 5;
    const int row0 = blockIdx.x * 64;            // = t * 64
    const int t    = blockIdx.x;
    const int m0   = wid * 16;
    const int col  = lane & 15, rhalf = (lane >> 4) * 8;
    const int m    = m0 + col;

    if (wid == 0) {                              // TDM ignores EXEC; one DMA per workgroup
        tdm_load_1d((unsigned)(uintptr_t)hstage, hinb + (size_t)row0 * 256, 64 * 256, 1);
        __builtin_amdgcn_s_wait_tensorcnt(0);
    }
    __syncthreads();

    f32x8 ar[4], az[4], ain[4], ahn[4];
#pragma unroll
    for (int rt = 0; rt < 4; ++rt) { ar[rt] = zero8(); az[rt] = zero8(); ain[rt] = zero8(); ahn[rt] = zero8(); }

    // gi = emb[a] @ w_ih_c^T  (K=32: one WMMA step; B reused across 4 row-tiles)
    {
        bf16x16 b_r = frag_b(wihc, 32, m0,       0, lane);
        bf16x16 b_z = frag_b(wihc, 32, 256 + m0, 0, lane);
        bf16x16 b_n = frag_b(wihc, 32, 512 + m0, 0, lane);
#pragma unroll
        for (int rt = 0; rt < 4; ++rt) {
            const int b    = rt * 16 + (lane & 15);
            const int aidx = acts[(40 + t + kstep) * 64 + b];
            bf16x16 a = frag_a_rowptr(embb + (size_t)aidx * 32, lane);
            ar[rt]  = wmma(a, b_r, ar[rt]);
            az[rt]  = wmma(a, b_z, az[rt]);
            ain[rt] = wmma(a, b_n, ain[rt]);
        }
    }
    // gh = h @ w_hh_c^T  (A from LDS; each B fragment reused for 4 row-tiles)
#pragma unroll
    for (int k = 0; k < 256; k += 32) {
        bf16x16 b_r = frag_b(whhc, 256, m0,       k, lane);
        bf16x16 b_z = frag_b(whhc, 256, 256 + m0, k, lane);
        bf16x16 b_n = frag_b(whhc, 256, 512 + m0, k, lane);
#pragma unroll
        for (int rt = 0; rt < 4; ++rt) {
            bf16x16 a = frag_a(hstage + rt * 16 * 272 + k, 272, lane);
            ar[rt]  = wmma(a, b_r, ar[rt]);
            az[rt]  = wmma(a, b_z, az[rt]);
            ahn[rt] = wmma(a, b_n, ahn[rt]);
        }
    }
    const float br  = bihc[m] + bhhc[m];
    const float bz  = bihc[256 + m] + bhhc[256 + m];
    const float bin = bihc[512 + m];
    const float bhn = bhhc[512 + m];
#pragma unroll
    for (int rt = 0; rt < 4; ++rt) {
#pragma unroll
        for (int v = 0; v < 8; ++v) {
            int rg = row0 + rt * 16 + v + rhalf;
            float r  = sigm(ar[rt][v] + br);
            float zz = sigm(az[rt][v] + bz);
            float nn = tanh_f(ain[rt][v] + bin + r * (ahn[rt][v] + bhn));
            float hp = hinf[(size_t)rg * 256 + m];
            float h2 = (1.0f - zz) * nn + zz * hp;
            houtf[(size_t)rg * 256 + m] = h2;
            houtb[(size_t)rg * 256 + m] = f2bf(h2);
        }
    }
}

// ------- open-loop MLP head: 64-row stripes, TDM-staged h, LDS-staged relu activations ---
__global__ __launch_bounds__(512) void k_open_mlp(int kstep,
                                                  const unsigned short* __restrict__ hb,
                                                  const unsigned short* __restrict__ w1b,
                                                  const float* __restrict__ b1,
                                                  const unsigned short* __restrict__ w2b,
                                                  const float* __restrict__ b2,
                                                  float* __restrict__ out) {
    __shared__ __attribute__((aligned(64))) unsigned short hstage[64 * 272];
    __shared__ __attribute__((aligned(64))) unsigned short a1[64 * 272];
    const int tid  = threadIdx.x, lane = tid & 31, wid = tid >> 5;
    const int row0 = blockIdx.x * 64;
    const int t    = blockIdx.x;
    const int col  = lane & 15, rhalf = (lane >> 4) * 8;

    if (wid == 0) {
        tdm_load_1d((unsigned)(uintptr_t)hstage, hb + (size_t)row0 * 256, 64 * 256, 1);
        __builtin_amdgcn_s_wait_tensorcnt(0);
    }
    __syncthreads();

    // phase 1: hidden layer (M=256), one col-tile per wave, 4 row-tiles
    {
        const int m0 = wid * 16, m = m0 + col;
        f32x8 acc[4];
#pragma unroll
        for (int rt = 0; rt < 4; ++rt) acc[rt] = zero8();
#pragma unroll
        for (int k = 0; k < 256; k += 32) {
            bf16x16 b = frag_b(w1b, 256, m0, k, lane);
#pragma unroll
            for (int rt = 0; rt < 4; ++rt)
                acc[rt] = wmma(frag_a(hstage + rt * 16 * 272 + k, 272, lane), b, acc[rt]);
        }
        float bias = b1[m];
#pragma unroll
        for (int rt = 0; rt < 4; ++rt)
#pragma unroll
            for (int v = 0; v < 8; ++v)
                a1[(rt * 16 + v + rhalf) * 272 + m] = f2bf(fmaxf(acc[rt][v] + bias, 0.0f));
    }
    __syncthreads();
    // phase 2: output layer (N=512), two col-tiles per wave, 4 row-tiles each
#pragma unroll
    for (int s = 0; s < 2; ++s) {
        const int n0 = (wid * 2 + s) * 16;
        f32x8 acc[4];
#pragma unroll
        for (int rt = 0; rt < 4; ++rt) acc[rt] = zero8();
#pragma unroll
        for (int k = 0; k < 256; k += 32) {
            bf16x16 b = frag_b(w2b, 256, n0, k, lane);
#pragma unroll
            for (int rt = 0; rt < 4; ++rt)
                acc[rt] = wmma(frag_a(a1 + rt * 16 * 272 + k, 272, lane), b, acc[rt]);
        }
        const int n = n0 + col;
        float bias = b2[n];
#pragma unroll
        for (int rt = 0; rt < 4; ++rt)
#pragma unroll
            for (int v = 0; v < 8; ++v) {
                size_t o = ((size_t)(t * 8 + kstep) * 64 + rt * 16 + v + rhalf) * 512 + n;
                out[o] = acc[rt][v] + bias;
            }
    }
}

// ---------------- host launcher ----------------
extern "C" void kernel_launch(void* const* d_in, const int* in_sizes, int n_in,
                              void* d_out, int out_size, void* d_ws, size_t ws_size,
                              hipStream_t stream) {
    const float* obs   = (const float*)d_in[0];
    const int*   acts  = (const int*)  d_in[1];
    const float* w_enc = (const float*)d_in[2];
    const float* b_enc = (const float*)d_in[3];
    const float* w_ih  = (const float*)d_in[4];
    const float* w_hh  = (const float*)d_in[5];
    const float* b_ih  = (const float*)d_in[6];
    const float* b_hh  = (const float*)d_in[7];
    const float* w_ihc = (const float*)d_in[8];
    const float* w_hhc = (const float*)d_in[9];
    const float* b_ihc = (const float*)d_in[10];
    const float* b_hhc = (const float*)d_in[11];
    const float* W1    = (const float*)d_in[12];
    const float* b1    = (const float*)d_in[13];
    const float* W2    = (const float*)d_in[14];
    const float* b2    = (const float*)d_in[15];
    const float* emb   = (const float*)d_in[16];
    float* out = (float*)d_out;

    char* ws = (char*)d_ws;
    size_t off = 0;
    auto alloc = [&](size_t bytes) { char* p = ws + off; off = (off + bytes + 255) & ~(size_t)255; return p; };
    unsigned short* wenc_bf = (unsigned short*)alloc((size_t)512 * 7072 * 2);
    unsigned short* wih_bf  = (unsigned short*)alloc((size_t)768 * 512 * 2);
    unsigned short* whh_bf  = (unsigned short*)alloc((size_t)768 * 256 * 2);
    unsigned short* wihc_bf = (unsigned short*)alloc((size_t)768 * 32 * 2);
    unsigned short* whhc_bf = (unsigned short*)alloc((size_t)768 * 256 * 2);
    unsigned short* W1_bf   = (unsigned short*)alloc((size_t)256 * 256 * 2);
    unsigned short* W2_bf   = (unsigned short*)alloc((size_t)512 * 256 * 2);
    unsigned short* emb_bf  = (unsigned short*)alloc((size_t)18 * 32 * 2);
    unsigned short* z_bf    = (unsigned short*)alloc((size_t)15360 * 512 * 2);
    float*          hA_f    = (float*)         alloc((size_t)12288 * 256 * 4);
    unsigned short* hA_b    = (unsigned short*)alloc((size_t)12288 * 256 * 2);
    float*          hB_f    = (float*)         alloc((size_t)12288 * 256 * 4);
    unsigned short* hB_b    = (unsigned short*)alloc((size_t)12288 * 256 * 2);
    (void)ws_size; (void)in_sizes; (void)n_in; (void)out_size;

    auto g = [](int n) { return dim3((n + 255) / 256); };
    k_cvt_pad<<<g(512 * 7072), 256, 0, stream>>>(w_enc, wenc_bf, 512, 7056, 7072);
    k_cvt<<<g(768 * 512), 256, 0, stream>>>(w_ih,  wih_bf,  768 * 512);
    k_cvt<<<g(768 * 256), 256, 0, stream>>>(w_hh,  whh_bf,  768 * 256);
    k_cvt<<<g(768 * 32),  256, 0, stream>>>(w_ihc, wihc_bf, 768 * 32);
    k_cvt<<<g(768 * 256), 256, 0, stream>>>(w_hhc, whhc_bf, 768 * 256);
    k_cvt<<<g(256 * 256), 256, 0, stream>>>(W1,    W1_bf,   256 * 256);
    k_cvt<<<g(512 * 256), 256, 0, stream>>>(W2,    W2_bf,   512 * 256);
    k_cvt<<<g(18 * 32),   256, 0, stream>>>(emb,   emb_bf,  18 * 32);

    k_encoder<<<240, 512, 0, stream>>>(obs, wenc_bf, b_enc, z_bf);
    k_gru_scan<<<4, 512, 0, stream>>>(z_bf, wih_bf, whh_bf, b_ih, b_hh, hA_f, hA_b);

    for (int k = 0; k < 8; ++k) {
        const float*          inf = (k & 1) ? hB_f : hA_f;
        const unsigned short* inb = (k & 1) ? hB_b : hA_b;
        float*                of  = (k & 1) ? hA_f : hB_f;
        unsigned short*       ob  = (k & 1) ? hA_b : hB_b;
        k_open_gru<<<192, 512, 0, stream>>>(k, acts, emb_bf, wihc_bf, whhc_bf,
                                            b_ihc, b_hhc, inf, inb, of, ob);
        k_open_mlp<<<192, 512, 0, stream>>>(k, ob, W1_bf, b1, W2_bf, b2, out);
    }
}
